// LanguageModel_15238543966867
// MI455X (gfx1250) — matrix operand order repaired and checked by reading, not compiled, not run on hardware
//
#include <hip/hip_runtime.h>
#include <hip/hip_bf16.h>
#include <math.h>

// ---------------- types for WMMA ----------------
typedef __attribute__((ext_vector_type(16))) __bf16 v16bf;
typedef __attribute__((ext_vector_type(8)))  __bf16 v8bf;
typedef __attribute__((ext_vector_type(8)))  float  v8f;

// ---------------- problem constants ----------------
#define BB 64
#define LL 30
#define TT 29          // L-1 time steps
#define EE 256
#define HH 1024
#define GG 4096        // 4*H
#define OO 512
#define VV 20000
#define M1 (BB*TT)     // 1856 rows

__device__ __forceinline__ __bf16 f32_to_bf16(float f) {
  unsigned u = __builtin_bit_cast(unsigned, f);
  unsigned r = u + 0x7FFFu + ((u >> 16) & 1u);   // round-to-nearest-even
  unsigned short h = (unsigned short)(r >> 16);
  return __builtin_bit_cast(__bf16, h);
}

__device__ __forceinline__ v16bf load_a_frag(const __bf16* arow, int k0, int half) {
  // A fragment, contiguous k as in B: elems 0..15 -> k = half*16 + 0..15
  return *(const v16bf*)(arow + k0 + half * 16);
}

// ---------------- prep: f32 (K x N) -> bf16 transposed (N x K) ----------------
__global__ void cvt_transpose_bf16(const float* __restrict__ src, __bf16* __restrict__ dst,
                                   int K, int N) {
  size_t idx = (size_t)blockIdx.x * blockDim.x + threadIdx.x;
  if (idx >= (size_t)K * N) return;
  int n = (int)(idx % N);
  int k = (int)(idx / N);
  dst[(size_t)n * K + k] = f32_to_bf16(src[idx]);
}

// ---------------- embedding gather -> bf16 xs (M1 x E) ----------------
__global__ void gather_embed(const int* __restrict__ ids, const float* __restrict__ emb,
                             __bf16* __restrict__ xs) {
  int idx = blockIdx.x * blockDim.x + threadIdx.x;   // exactly M1*EE threads
  int r = idx >> 8;            // row in [0, M1)
  int e = idx & (EE - 1);
  int b = r / TT;
  int t = r - b * TT;
  int id = ids[b * LL + t];
  xs[idx] = f32_to_bf16(emb[(size_t)id * EE + e]);
}

// ---------------- zero LSTM state ----------------
__global__ void lstm_init(float* __restrict__ c, __bf16* __restrict__ h0) {
  int idx = blockIdx.x * blockDim.x + threadIdx.x;   // BB*HH threads
  c[idx] = 0.0f;
  h0[idx] = f32_to_bf16(0.0f);
}

// ---------------- register-blocked WMMA GEMM ----------------
// Each wave computes a (16*MT) x (16*NT) output tile.
// D = A(M x K bf16 row-major) * BT(N x K bf16)^T [+ bias];  optional bf16 mirror.
// Requires M % (16*MT) == 0, N % (16*NT) == 0, K % 32 == 0.
template <int MT, int NT>
__global__ void wmma_gemm(const __bf16* __restrict__ A, const __bf16* __restrict__ BT,
                          const float* __restrict__ bias,
                          float* __restrict__ D, __bf16* __restrict__ Dbf,
                          int M, int N, int K, int d_stride) {
  int wave   = blockIdx.x * (blockDim.x >> 5) + (threadIdx.x >> 5);
  int tilesN = N / (16 * NT);
  int total  = (M / (16 * MT)) * tilesN;
  if (wave >= total) return;                 // uniform per wave -> EXEC all-ones inside
  int tm = wave / tilesN;
  int tn = wave - tm * tilesN;
  int lane = threadIdx.x & 31;
  int half = lane >> 4;
  int mn   = lane & 15;

  const __bf16* arow[MT];
  const __bf16* brow[NT];
#pragma unroll
  for (int mi = 0; mi < MT; ++mi)
    arow[mi] = A + (size_t)(tm * 16 * MT + mi * 16 + mn) * K;
#pragma unroll
  for (int ni = 0; ni < NT; ++ni)
    brow[ni] = BT + (size_t)(tn * 16 * NT + ni * 16 + mn) * K;

  v8f acc[MT][NT] = {};
  for (int k0 = 0; k0 < K; k0 += 32) {
    v16bf a[MT], b[NT];
#pragma unroll
    for (int mi = 0; mi < MT; ++mi) a[mi] = load_a_frag(arow[mi], k0, half);
#pragma unroll
    for (int ni = 0; ni < NT; ++ni)
      b[ni] = *(const v16bf*)(brow[ni] + k0 + half * 16);   // k = half*16 + e
#pragma unroll
    for (int mi = 0; mi < MT; ++mi)
#pragma unroll
      for (int ni = 0; ni < NT; ++ni)
        acc[mi][ni] = __builtin_amdgcn_wmma_f32_16x16x32_bf16(
            false, a[mi], false, b[ni], (short)0, acc[mi][ni], false, false);
  }
#pragma unroll
  for (int ni = 0; ni < NT; ++ni) {
    int col = tn * 16 * NT + ni * 16 + mn;
    float bv = bias ? bias[col] : 0.0f;
#pragma unroll
    for (int mi = 0; mi < MT; ++mi) {
#pragma unroll
      for (int r = 0; r < 8; ++r) {
        int row = tm * 16 * MT + mi * 16 + half * 8 + r;   // f32 C/D layout
        float v = acc[mi][ni][r] + bv;
        D[(size_t)row * d_stride + col] = v;
        if (Dbf) Dbf[(size_t)row * d_stride + col] = f32_to_bf16(v);
      }
    }
  }
}

// ---------------- fully fused LSTM step ----------------
// One kernel per time step. Each wave owns 16 j-columns across ALL FOUR gate
// blocks (N-subtiles of Wr at stride H), so after the K-loop it holds
// zi/zf/zg/zo in registers, adds zx[:,t,:], applies the gates, updates c and
// writes h_t (bf16, ping-pong) plus hs[:,t,:].
template <int MT>
__global__ void lstm_step_fused(const __bf16* __restrict__ hprev,
                                const __bf16* __restrict__ WrT,   // (4H x H) bf16
                                const float*  __restrict__ zxt,   // zx + t*GG, row stride TT*GG
                                float*  __restrict__ c,
                                __bf16* __restrict__ hnext,
                                __bf16* __restrict__ hsb, int t) {
  int wave   = blockIdx.x * (blockDim.x >> 5) + (threadIdx.x >> 5);
  int tilesJ = HH / 16;                         // 64
  int total  = (BB / (16 * MT)) * tilesJ;
  if (wave >= total) return;
  int tm = wave / tilesJ;
  int tj = wave - tm * tilesJ;
  int lane = threadIdx.x & 31;
  int half = lane >> 4;
  int mn   = lane & 15;

  const __bf16* arow[MT];
#pragma unroll
  for (int mi = 0; mi < MT; ++mi)
    arow[mi] = hprev + (size_t)(tm * 16 * MT + mi * 16 + mn) * HH;
  const __bf16* brow[4];
#pragma unroll
  for (int g = 0; g < 4; ++g)
    brow[g] = WrT + (size_t)(g * HH + tj * 16 + mn) * HH;

  v8f acc[MT][4] = {};
  for (int k0 = 0; k0 < HH; k0 += 32) {
    v16bf a[MT], b[4];
#pragma unroll
    for (int mi = 0; mi < MT; ++mi) a[mi] = load_a_frag(arow[mi], k0, half);
#pragma unroll
    for (int g = 0; g < 4; ++g)
      b[g] = *(const v16bf*)(brow[g] + k0 + half * 16);
#pragma unroll
    for (int mi = 0; mi < MT; ++mi)
#pragma unroll
      for (int g = 0; g < 4; ++g)
        acc[mi][g] = __builtin_amdgcn_wmma_f32_16x16x32_bf16(
            false, a[mi], false, b[g], (short)0, acc[mi][g], false, false);
  }

  int j = tj * 16 + mn;
#pragma unroll
  for (int mi = 0; mi < MT; ++mi) {
#pragma unroll
    for (int r = 0; r < 8; ++r) {
      int row = tm * 16 * MT + mi * 16 + half * 8 + r;     // batch index b
      const float* zr = zxt + (size_t)row * ((size_t)TT * GG);
      float zi = acc[mi][0][r] + zr[j];
      float zf = acc[mi][1][r] + zr[HH + j];
      float zg = acc[mi][2][r] + zr[2 * HH + j];
      float zo = acc[mi][3][r] + zr[3 * HH + j];
      float ig = fminf(fmaxf(0.2f * zi + 0.5f, 0.0f), 1.0f);
      float fg = fminf(fmaxf(0.2f * zf + 0.5f, 0.0f), 1.0f);
      float og = fminf(fmaxf(0.2f * zo + 0.5f, 0.0f), 1.0f);
      size_t cidx = (size_t)row * HH + j;
      float cc = fg * c[cidx] + ig * tanhf(zg);
      float hh = og * tanhf(cc);
      c[cidx] = cc;
      __bf16 hb = f32_to_bf16(hh);
      hnext[cidx] = hb;
      hsb[((size_t)row * TT + t) * HH + j] = hb;
    }
  }
}

// ---------------- softmax over vocab, one block per row ----------------
__global__ void softmax_rows(const float* __restrict__ logits, float* __restrict__ preds) {
  int row = blockIdx.x;
  const float* x = logits + (size_t)row * VV;
  float* y = preds + (size_t)row * VV;
  __shared__ float red[256];
  int tid = threadIdx.x;
  float m = -INFINITY;
  for (int i = tid; i < VV; i += 256) m = fmaxf(m, x[i]);
  red[tid] = m; __syncthreads();
  for (int s = 128; s > 0; s >>= 1) {
    if (tid < s) red[tid] = fmaxf(red[tid], red[tid + s]);
    __syncthreads();
  }
  m = red[0]; __syncthreads();
  float sum = 0.0f;
  for (int i = tid; i < VV; i += 256) { float e = __expf(x[i] - m); y[i] = e; sum += e; }
  red[tid] = sum; __syncthreads();
  for (int s = 128; s > 0; s >>= 1) {
    if (tid < s) red[tid] += red[tid + s];
    __syncthreads();
  }
  float inv = 1.0f / red[0];
  for (int i = tid; i < VV; i += 256) y[i] *= inv;
}

// ---------------- host side ----------------
extern "C" void kernel_launch(void* const* d_in, const int* in_sizes, int n_in,
                              void* d_out, int out_size, void* d_ws, size_t ws_size,
                              hipStream_t stream) {
  (void)in_sizes; (void)n_in; (void)out_size; (void)ws_size;
  const int*   ids = (const int*)d_in[0];
  const float* emb = (const float*)d_in[1];
  const float* Wk  = (const float*)d_in[2];
  const float* Wr  = (const float*)d_in[3];
  const float* bg  = (const float*)d_in[4];
  const float* Wp  = (const float*)d_in[5];
  const float* bp  = (const float*)d_in[6];
  const float* Ws  = (const float*)d_in[7];
  float* logits = (float*)d_out;
  float* preds  = logits + (size_t)M1 * VV;

  char* p = (char*)d_ws;
  auto alloc = [&](size_t bytes) -> void* {
    void* r = (void*)p;
    p += (bytes + 255) & ~(size_t)255;
    return r;
  };
  __bf16* WkT  = (__bf16*)alloc((size_t)GG * EE * 2);   // 4096 x 256
  __bf16* WrT  = (__bf16*)alloc((size_t)GG * HH * 2);   // 4096 x 1024
  __bf16* WpT  = (__bf16*)alloc((size_t)OO * HH * 2);   // 512 x 1024
  __bf16* WsT  = (__bf16*)alloc((size_t)VV * OO * 2);   // 20000 x 512
  __bf16* xs   = (__bf16*)alloc((size_t)M1 * EE * 2);   // 1856 x 256
  float*  zx   = (float*) alloc((size_t)M1 * GG * 4);   // 1856 x 4096
  float*  c    = (float*) alloc((size_t)BB * HH * 4);
  __bf16* hbf0 = (__bf16*)alloc((size_t)BB * HH * 2);   // h ping
  __bf16* hbf1 = (__bf16*)alloc((size_t)BB * HH * 2);   // h pong
  __bf16* hsb  = (__bf16*)alloc((size_t)M1 * HH * 2);   // 1856 x 1024
  float*  outf = (float*) alloc((size_t)M1 * OO * 4);
  __bf16* outb = (__bf16*)alloc((size_t)M1 * OO * 2);

  // 1) weight prep: f32 KxN -> bf16 NxK
  cvt_transpose_bf16<<<(EE * GG + 255) / 256, 256, 0, stream>>>(Wk, WkT, EE, GG);
  cvt_transpose_bf16<<<(HH * GG + 255) / 256, 256, 0, stream>>>(Wr, WrT, HH, GG);
  cvt_transpose_bf16<<<(HH * OO + 255) / 256, 256, 0, stream>>>(Wp, WpT, HH, OO);
  cvt_transpose_bf16<<<((size_t)OO * VV + 255) / 256, 256, 0, stream>>>(Ws, WsT, OO, VV);

  // 2) embedding gather
  gather_embed<<<(M1 * EE) / 256, 256, 0, stream>>>(ids, emb, xs);

  // 3) zx = xs @ Wk + b   (1856x4096, K=256): 64x32 tiles -> 29*128 = 3712 waves
  wmma_gemm<4, 2><<<3712 / 4, 128, 0, stream>>>(
      xs, WkT, bg, zx, nullptr, M1, GG, EE, GG);

  // 4) LSTM recurrence: one fused kernel per step (GEMM + gates in registers)
  lstm_init<<<(BB * HH) / 256, 256, 0, stream>>>(c, hbf0);
  __bf16* hp = hbf0;
  __bf16* hn = hbf1;
  for (int t = 0; t < TT; ++t) {
    // 32x(16 j-cols x 4 gates) tiles -> (64/32)*64 = 128 waves
    lstm_step_fused<2><<<128 / 4, 128, 0, stream>>>(
        hp, WrT, zx + (size_t)t * GG, c, hn, hsb, t);
    __bf16* tmp = hp; hp = hn; hn = tmp;
  }

  // 5) out = hs @ Wp + bp   (1856x512, K=1024): 29*16 = 464 waves; also emit bf16
  wmma_gemm<4, 2><<<464 / 4, 128, 0, stream>>>(
      hsb, WpT, bp, outf, outb, M1, OO, HH, OO);

  // 6) logits = out @ Ws   (1856x20000, K=512): 29*625 = 18125 waves, into d_out
  wmma_gemm<4, 2><<<(18125 + 3) / 4, 128, 0, stream>>>(
      outb, WsT, nullptr, logits, nullptr, M1, VV, OO, VV);

  // 7) preds = softmax(logits)
  softmax_rows<<<M1, 256, 0, stream>>>(logits, preds);
}
